// PanODE_23725399343241
// MI455X (gfx1250) — matrix operand order repaired and checked by reading, not compiled, not run on hardware
//
#include <hip/hip_runtime.h>
#include <hip/hip_bf16.h>
#include <math.h>

typedef __attribute__((ext_vector_type(16))) __bf16 v16bf;
typedef __attribute__((ext_vector_type(8)))  float  v8f;

#define BATCH 512
#define DIM   256
#define HID   512
#define NC    32      // NUM_COEFF
#define NT    30      // NC-2
#define NP    64      // Chebyshev points
#define TEV   16
#define MAXIT 10
#define TOLC  0.01f

// ---- spectral/state block layout (float offsets within ws) ----
#define S_TCHEB  0        // 64
#define S_PHI    64       // 32*64
#define S_PD2    2112     // 30*2  [j][k]
#define S_INV0   2172     // [2][2] row-major
#define S_M2     2176     // [2][30]
#define S_PTQ    2240     // [64][30]
#define S_PHIOUT 4160     // [32][16]
#define S_T0     4672
#define S_SUMSQ  4673
#define S_DELTA  4674
#define S_DONE   4675     // int
#define SPECT_BYTES 32768

__device__ __forceinline__ unsigned short f2bf(float x) {
  union { float f; unsigned u; } c; c.f = x;
  unsigned r = c.u + 0x7FFFu + ((c.u >> 16) & 1u);
  return (unsigned short)(r >> 16);
}

// =============== spectral operator setup (1 block, 64 threads) ===============
__global__ void setup_spectral(const float* __restrict__ t_eval,
                               float* __restrict__ S,
                               float* __restrict__ out) {
  __shared__ float sPhi[NC * NP];
  __shared__ float sDPhi[NC * NP];
  __shared__ float sPhib[NT * NP];
  __shared__ float sDt[NP];
  __shared__ float sAug[NT][2 * NT];
  __shared__ float sPTQ[NP * NT];
  __shared__ float sInv0[4];
  __shared__ float sPD2[NT * 2];

  const int p = threadIdx.x;      // 64 threads
  const float PI = 3.14159265358979323846f;
  float t0 = t_eval[0], t1 = t_eval[TEV - 1];
  float sgn = (t1 > t0) ? 1.f : ((t1 < t0) ? -1.f : 0.f);
  float tc  = -sgn * cosf(PI * (float)p / (float)NP);
  float tcn = (p == NP - 1) ? 1.0f : -sgn * cosf(PI * (float)(p + 1) / (float)NP);
  sDt[p] = tcn - tc;
  S[S_TCHEB + p] = tc;

  // Chebyshev T_n column
  { float a = 1.f, b = tc;
    sPhi[0 * NP + p] = 1.f; sPhi[1 * NP + p] = tc;
    for (int n = 2; n < NC; ++n) { float v = 2.f * tc * b - a; sPhi[n * NP + p] = v; a = b; b = v; } }
  // DPhi = (2/(t1-t0)) * k*U_{k-1}
  { float fac = 2.f / (t1 - t0);
    float um2 = 1.f, um1 = 2.f * tc;
    sDPhi[0 * NP + p] = 0.f;
    sDPhi[1 * NP + p] = fac * 1.f;
    sDPhi[2 * NP + p] = fac * 2.f * um1;
    for (int n = 3; n < NC; ++n) { float u = 2.f * tc * um1 - um2; um2 = um1; um1 = u;
                                   sDPhi[n * NP + p] = fac * (float)n * u; } }
  __syncthreads();

  if (p == 0) {
    float a = sPhi[0], b = sDPhi[0], c = sPhi[NP], d = sDPhi[NP];
    float id = 1.f / (a * d - b * c);
    sInv0[0] =  d * id; sInv0[1] = -b * id;
    sInv0[2] = -c * id; sInv0[3] =  a * id;
    S[S_T0] = t0; S[S_SUMSQ] = 0.f; S[S_DELTA] = __builtin_huge_valf();
    ((int*)S)[S_DONE] = 0;
  }
  if (p < NT * 2) { int j = p >> 1, k = p & 1;
                    sPD2[p] = k ? sDPhi[(2 + j) * NP] : sPhi[(2 + j) * NP]; }
  __syncthreads();

  // Phi_b[j][p] = DPhi[2+j][p] - (PD2@inv0)[j][:] . D01[:][p]
  for (int j = 0; j < NT; ++j) {
    float c0 = sPD2[j * 2] * sInv0[0] + sPD2[j * 2 + 1] * sInv0[2];
    float c1 = sPD2[j * 2] * sInv0[1] + sPD2[j * 2 + 1] * sInv0[3];
    sPhib[j * NP + p] = sDPhi[(2 + j) * NP + p] - c0 * sDPhi[0 * NP + p] - c1 * sDPhi[1 * NP + p];
  }
  __syncthreads();

  // G = Phi_b @ (Dt*Phi_b^T) into Aug left; identity right
  for (int idx = p; idx < NT * NT; idx += NP) {
    int i = idx / NT, j = idx % NT;
    float s = 0.f;
    for (int q = 0; q < NP; ++q) s += sPhib[i * NP + q] * sDt[q] * sPhib[j * NP + q];
    sAug[i][j] = s;
  }
  for (int idx = p; idx < NT * NT; idx += NP) {
    int i = idx / NT, j = idx % NT;
    sAug[i][NT + j] = (i == j) ? 1.f : 0.f;
  }
  __syncthreads();

  // Gauss-Jordan (G is SPD -> no pivoting)
  for (int k = 0; k < NT; ++k) {
    float ip = 1.f / sAug[k][k];
    __syncthreads();
    if (p < 2 * NT) sAug[k][p] *= ip;
    __syncthreads();
    if (p < NT && p != k) {
      float f = sAug[p][k];
      for (int j = 0; j < 2 * NT; ++j) sAug[p][j] -= f * sAug[k][j];
    }
    __syncthreads();
  }

  // PhibTQ[p][n] = Dt[p] * sum_j Phib[j][p]*Q[j][n]
  for (int n = 0; n < NT; ++n) {
    float s = 0.f;
    for (int j = 0; j < NT; ++j) s += sPhib[j * NP + p] * sAug[j][NT + n];
    float v = sDt[p] * s;
    sPTQ[p * NT + n] = v;
    S[S_PTQ + p * NT + n] = v;
  }
  __syncthreads();

  // M2[k][n] = sum_p (inv0@D01)[k][p] * PhibTQ[p][n]
  if (p < NT) {
    for (int k = 0; k < 2; ++k) {
      float s = 0.f;
      for (int q = 0; q < NP; ++q) {
        float row = sInv0[k * 2] * sDPhi[0 * NP + q] + sInv0[k * 2 + 1] * sDPhi[1 * NP + q];
        s += row * sPTQ[q * NT + p];
      }
      S[S_M2 + k * NT + p] = s;
    }
  }
  for (int idx = p; idx < NC * NP; idx += NP) S[S_PHI + idx] = sPhi[idx];
  if (p < NT * 2) S[S_PD2 + p] = sPD2[p];
  if (p < 4)      S[S_INV0 + p] = sInv0[p];
  if (p < TEV) {
    float te = t_eval[p];
    out[p] = te;   // t_eval output segment
    float to = -1.f + 2.f * (te - t0) / (t1 - t0);
    float a = 1.f, b = to;
    S[S_PHIOUT + 0 * TEV + p] = 1.f;
    S[S_PHIOUT + 1 * TEV + p] = to;
    for (int n = 2; n < NC; ++n) { float v = 2.f * to * b - a; S[S_PHIOUT + n * TEV + p] = v; a = b; b = v; }
  }
}

// =============== weight packing into WMMA B-fragment layout ===============
// dst[((kb*(N/16)+nb)*32 + lane)*16 + e] = bf16(W[k][n]),
// n = nb*16 + (lane&15), k = kb*32 + (lane>>4)*16 + e
__global__ void pack_weights(const float* __restrict__ W, unsigned short* __restrict__ Wp,
                             int K, int N) {
  int idx = blockIdx.x * blockDim.x + threadIdx.x;
  if (idx >= K * N) return;
  int e    = idx & 15;
  int lane = (idx >> 4) & 31;
  int rest = idx >> 9;
  int nb   = rest % (N >> 4);
  int kb   = rest / (N >> 4);
  int n = (nb << 4) + (lane & 15);
  int k = (kb << 5) + ((lane >> 4) << 4) + e;
  Wp[idx] = f2bf(W[(size_t)k * N + n]);
}

__global__ void cvt_bf16(const float* __restrict__ x, unsigned short* __restrict__ y, int n) {
  int i = blockIdx.x * blockDim.x + threadIdx.x;
  if (i < n) y[i] = f2bf(x[i]);
}

// =============== bf16 WMMA GEMM: C[M,N] = A[M,K] * B[K,N] (+bias, +t, tanh) ===============
// block = 8 waves, wave -> 16x64 tile (4 accumulators), block tile 128x64
// mode 0: out f32 = acc + bias
// mode 1: out bf16 = tanh(acc + bias + t0)
// mode 2: out bf16 = tanh(acc + bias + t_cheb[row & 63])
union ABf { uint4 q[2]; v16bf v; };

__global__ __launch_bounds__(256) void wmma_gemm(
    const unsigned short* __restrict__ A,
    const unsigned short* __restrict__ Bp,
    const float* __restrict__ bias,
    const float* __restrict__ Sv,
    int K, int N, int mode,
    unsigned short* __restrict__ obf,
    float* __restrict__ of) {
  const int lane  = threadIdx.x & 31;
  const int wave  = threadIdx.x >> 5;
  const int m0    = blockIdx.x * 128 + wave * 16;
  const int nbase = blockIdx.y * 64;
  const int mrow  = lane & 15;
  const int khalf = lane >> 4;
  const int nblk  = N >> 4;

  v8f acc[4] = {};
  const unsigned short* Arow = A + (size_t)(m0 + mrow) * K + khalf * 16;
  const unsigned short* Bbase = Bp + ((size_t)(nbase >> 4) * 32 + lane) * 16;
  const size_t bKstep = (size_t)nblk * 32 * 16;   // ushorts per kb step

  #pragma unroll 4
  for (int k0 = 0; k0 < K; k0 += 32) {
    ABf a;
    a.q[0] = *(const uint4*)(Arow + k0);
    a.q[1] = *(const uint4*)(Arow + k0 + 8);
    const unsigned short* bp = Bbase + (size_t)(k0 >> 5) * bKstep;
    __builtin_prefetch(Arow + k0 + 32, 0, 1);
    __builtin_prefetch(bp + bKstep, 0, 1);
    ABf b0, b1, b2, b3;
    b0.q[0] = *(const uint4*)(bp);         b0.q[1] = *(const uint4*)(bp + 8);
    b1.q[0] = *(const uint4*)(bp + 512);   b1.q[1] = *(const uint4*)(bp + 520);
    b2.q[0] = *(const uint4*)(bp + 1024);  b2.q[1] = *(const uint4*)(bp + 1032);
    b3.q[0] = *(const uint4*)(bp + 1536);  b3.q[1] = *(const uint4*)(bp + 1544);
    acc[0] = __builtin_amdgcn_wmma_f32_16x16x32_bf16(false, a.v, false, b0.v, (short)0, acc[0], false, false);
    acc[1] = __builtin_amdgcn_wmma_f32_16x16x32_bf16(false, a.v, false, b1.v, (short)0, acc[1], false, false);
    acc[2] = __builtin_amdgcn_wmma_f32_16x16x32_bf16(false, a.v, false, b2.v, (short)0, acc[2], false, false);
    acc[3] = __builtin_amdgcn_wmma_f32_16x16x32_bf16(false, a.v, false, b3.v, (short)0, acc[3], false, false);
  }

  const float tscal = (mode == 1) ? Sv[S_T0] : 0.f;
  const int coll = lane & 15;
  for (int j = 0; j < 4; ++j) {
    int n = nbase + j * 16 + coll;
    float bn = bias[n];
    #pragma unroll
    for (int r = 0; r < 8; ++r) {
      int row = m0 + khalf * 8 + r;
      float v = acc[j][r] + bn;
      if (mode == 1) v = tanhf(v + tscal);
      else if (mode == 2) v = tanhf(v + Sv[S_TCHEB + (row & 63)]);
      if (obf) obf[(size_t)row * N + n] = f2bf(v);
      else     of[(size_t)row * N + n] = v;
    }
  }
}

// =============== head coefficients + y_grid (bf16, GEMM1-A layout) ===============
__global__ __launch_bounds__(256) void head_ygrid(
    const float* __restrict__ B, const float* __restrict__ y0,
    const float* __restrict__ f0, const float* __restrict__ S,
    unsigned short* __restrict__ ybf) {
  __shared__ float sPhi[NC * NP];
  const int b = blockIdx.x, d = threadIdx.x;
  for (int i = d; i < NC * NP; i += 256) sPhi[i] = S[S_PHI + i];
  __syncthreads();
  float Bt[NT];
  const float* brow = B + ((size_t)b * DIM + d) * NT;
  #pragma unroll
  for (int j = 0; j < NT; ++j) Bt[j] = brow[j];
  float r0 = y0[b * DIM + d], r1 = f0[b * DIM + d];
  #pragma unroll
  for (int j = 0; j < NT; ++j) { r0 -= Bt[j] * S[S_PD2 + j * 2]; r1 -= Bt[j] * S[S_PD2 + j * 2 + 1]; }
  float i00 = S[S_INV0], i01 = S[S_INV0 + 1], i10 = S[S_INV0 + 2], i11 = S[S_INV0 + 3];
  float a0 = r0 * i00 + r1 * i10;
  float a1 = r0 * i01 + r1 * i11;
  for (int p = 0; p < NP; ++p) {
    float acc = a0 * sPhi[p] + a1 * sPhi[NP + p];
    #pragma unroll
    for (int n = 2; n < NC; ++n) acc += Bt[n - 2] * sPhi[n * NP + p];
    ybf[((size_t)b * NP + p) * DIM + d] = f2bf(acc);
  }
}

// =============== B update: fap@PhibTQ - yf2@M2, masked write, Frobenius norm ===============
__global__ __launch_bounds__(256) void bnew_kernel(
    const float* __restrict__ fap, const float* __restrict__ Bprev,
    const float* __restrict__ y0, const float* __restrict__ f0,
    float* __restrict__ S, float* __restrict__ Bnext) {
  __shared__ float sPTQ[NP * NT];
  __shared__ float red[256];
  const int b = blockIdx.x, d = threadIdx.x;
  for (int i = d; i < NP * NT; i += 256) sPTQ[i] = S[S_PTQ + i];
  __syncthreads();
  float yv = y0[b * DIM + d], fv = f0[b * DIM + d];
  float acc[NT];
  #pragma unroll
  for (int n = 0; n < NT; ++n) acc[n] = -(yv * S[S_M2 + n] + fv * S[S_M2 + NT + n]);
  for (int p = 0; p < NP; ++p) {
    float fp = fap[((size_t)b * NP + p) * DIM + d];
    #pragma unroll
    for (int n = 0; n < NT; ++n) acc[n] += fp * sPTQ[p * NT + n];
  }
  const int done = ((const int*)S)[S_DONE];
  const float* bp = Bprev + ((size_t)b * DIM + d) * NT;
  float* bn = Bnext + ((size_t)b * DIM + d) * NT;
  float sq = 0.f;
  #pragma unroll
  for (int n = 0; n < NT; ++n) {
    float pv = bp[n];
    float df = acc[n] - pv;
    sq += df * df;
    bn[n] = done ? pv : acc[n];
  }
  red[d] = sq; __syncthreads();
  for (int s = 128; s > 0; s >>= 1) { if (d < s) red[d] += red[d + s]; __syncthreads(); }
  if (d == 0) atomicAdd(&S[S_SUMSQ], red[0]);
}

__global__ void finalize_iter(float* S) {
  float d = sqrtf(S[S_SUMSQ]);
  int* dn = (int*)S + S_DONE;
  if (!(*dn)) {
    S[S_DELTA] = d;
    if (d < TOLC) *dn = 1;
  }
  S[S_SUMSQ] = 0.f;
}

// =============== trajectory projection ===============
__global__ __launch_bounds__(256) void traj_kernel(
    const float* __restrict__ B, const float* __restrict__ y0,
    const float* __restrict__ f0, const float* __restrict__ S,
    float* __restrict__ out) {
  __shared__ float sPO[NC * TEV];
  const int b = blockIdx.x, d = threadIdx.x;
  for (int i = d; i < NC * TEV; i += 256) sPO[i] = S[S_PHIOUT + i];
  __syncthreads();
  float Bt[NT];
  const float* brow = B + ((size_t)b * DIM + d) * NT;
  #pragma unroll
  for (int j = 0; j < NT; ++j) Bt[j] = brow[j];
  float r0 = y0[b * DIM + d], r1 = f0[b * DIM + d];
  #pragma unroll
  for (int j = 0; j < NT; ++j) { r0 -= Bt[j] * S[S_PD2 + j * 2]; r1 -= Bt[j] * S[S_PD2 + j * 2 + 1]; }
  float i00 = S[S_INV0], i01 = S[S_INV0 + 1], i10 = S[S_INV0 + 2], i11 = S[S_INV0 + 3];
  float a0 = r0 * i00 + r1 * i10;
  float a1 = r0 * i01 + r1 * i11;
  for (int t = 0; t < TEV; ++t) {
    float acc = a0 * sPO[t] + a1 * sPO[TEV + t];
    #pragma unroll
    for (int n = 2; n < NC; ++n) acc += Bt[n - 2] * sPO[n * TEV + t];
    out[(size_t)t * BATCH * DIM + b * DIM + d] = acc;
  }
}

__global__ void copyf(const float* __restrict__ src, float* __restrict__ dst, int n) {
  for (int i = blockIdx.x * blockDim.x + threadIdx.x; i < n; i += gridDim.x * blockDim.x)
    dst[i] = src[i];
}

__global__ void write_delta(const float* __restrict__ S, float* __restrict__ out) {
  out[0] = S[S_DELTA];
}

extern "C" void kernel_launch(void* const* d_in, const int* in_sizes, int n_in,
                              void* d_out, int out_size, void* d_ws, size_t ws_size,
                              hipStream_t stream) {
  const float* y_init = (const float*)d_in[0];
  const float* t_eval = (const float*)d_in[1];
  const float* B_init = (const float*)d_in[2];
  const float* W1     = (const float*)d_in[3];
  const float* b1     = (const float*)d_in[4];
  const float* W2     = (const float*)d_in[5];
  const float* b2     = (const float*)d_in[6];
  float* out = (float*)d_out;

  char* ws = (char*)d_ws;
  float*          S    = (float*)ws;
  float*          fini = (float*)(ws + 32768);
  unsigned short* W1p  = (unsigned short*)(ws + 557056);
  unsigned short* W2p  = (unsigned short*)(ws + 819200);
  unsigned short* ybf  = (unsigned short*)(ws + 1081344);
  unsigned short* hbf  = (unsigned short*)(ws + 17858560ull);
  float*          fap  = (float*)(ws + 51412992ull);
  float*          B0   = (float*)(ws + 84967424ull);
  float*          B1   = (float*)(ws + 100696064ull);

  setup_spectral<<<1, 64, 0, stream>>>(t_eval, S, out);
  pack_weights<<<(DIM * HID + 255) / 256, 256, 0, stream>>>(W1, W1p, DIM, HID);
  pack_weights<<<(HID * DIM + 255) / 256, 256, 0, stream>>>(W2, W2p, HID, DIM);
  cvt_bf16<<<(BATCH * DIM + 255) / 256, 256, 0, stream>>>(y_init, ybf, BATCH * DIM);

  // f_init = vf_point(t0, y_init): two WMMA GEMMs, M=512
  wmma_gemm<<<dim3(BATCH / 128, HID / 64), 256, 0, stream>>>(ybf, W1p, b1, S, DIM, HID, 1, hbf, nullptr);
  wmma_gemm<<<dim3(BATCH / 128, DIM / 64), 256, 0, stream>>>(hbf, W2p, b2, S, HID, DIM, 0, nullptr, fini);

  hipMemcpyAsync(B0, B_init, (size_t)BATCH * DIM * NT * sizeof(float),
                 hipMemcpyDeviceToDevice, stream);

  float* Bc = B0; float* Bn = B1;
  for (int it = 0; it < MAXIT; ++it) {
    head_ygrid<<<BATCH, 256, 0, stream>>>(Bc, y_init, fini, S, ybf);
    wmma_gemm<<<dim3(BATCH * NP / 128, HID / 64), 256, 0, stream>>>(ybf, W1p, b1, S, DIM, HID, 2, hbf, nullptr);
    wmma_gemm<<<dim3(BATCH * NP / 128, DIM / 64), 256, 0, stream>>>(hbf, W2p, b2, S, HID, DIM, 0, nullptr, fap);
    bnew_kernel<<<BATCH, 256, 0, stream>>>(fap, Bc, y_init, fini, S, Bn);
    finalize_iter<<<1, 1, 0, stream>>>(S);
    float* t = Bc; Bc = Bn; Bn = t;
  }

  // outputs: [0,16) t_eval (written in setup) | [16, 16+16*512*256) traj | delta | B
  traj_kernel<<<BATCH, 256, 0, stream>>>(Bc, y_init, fini, S, out + 16);
  write_delta<<<1, 1, 0, stream>>>(S, out + 16 + TEV * BATCH * DIM);
  copyf<<<2048, 256, 0, stream>>>(Bc, out + 16 + TEV * BATCH * DIM + 1, BATCH * DIM * NT);
}
